// Transformer_45947560132736
// MI455X (gfx1250) — matrix operand order repaired and checked by reading, not compiled, not run on hardware
//
#include <hip/hip_runtime.h>
#include <hip/hip_bf16.h>
#include <math.h>

typedef __bf16 bf16;
typedef __attribute__((ext_vector_type(16))) __bf16 bf16x16;
typedef __attribute__((ext_vector_type(8)))  float  f32x8;
typedef unsigned int v4u __attribute__((ext_vector_type(4)));
typedef int v8i __attribute__((ext_vector_type(8)));
typedef int v4i __attribute__((ext_vector_type(4)));

constexpr int Bx = 2, Hh = 8, Nn = 1024, Dm = 512, Dh = 64, Ll = 4;
constexpr float SLOPE = 0.01f;
constexpr float ATTN_SCALE = 0.044194173824159216f; // 512^-0.5

#if defined(__gfx1250__) && __has_builtin(__builtin_amdgcn_tensor_load_to_lds) && __has_builtin(__builtin_amdgcn_s_wait_tensorcnt)
#define HAVE_TDM 1
#else
#define HAVE_TDM 0
#endif

// ---------------------------------------------------------------------------
// WMMA helpers (CDNA5 16x16x32 bf16, wave32). Acc layout per ISA 7.12.2; K order here: k=16*(l>>4)+s.
// ---------------------------------------------------------------------------
__device__ __forceinline__ f32x8 zero8() {
    f32x8 v;
#pragma unroll
    for (int i = 0; i < 8; ++i) v[i] = 0.f;
    return v;
}

__device__ __forceinline__ f32x8 wmma_bf16(bf16x16 a, bf16x16 b, f32x8 c) {
    return __builtin_amdgcn_wmma_f32_16x16x32_bf16(
        false, a, false, b, (short)0, c, false, false);
}

// A fragment: 16x32 (MxK), row-major LDS tile. Lane l: row M=l&15, K chunk
// [kb..kb+15] with kb=(l>>4)*16 -> two contiguous 16B loads.
__device__ __forceinline__ bf16x16 load_A_frag(const bf16* smem, int row_stride) {
    int lane = threadIdx.x & 31;
    int m = lane & 15, kb = (lane >> 4) * 16;
    const bf16* p = smem + m * row_stride + kb;
    union { bf16x16 v; uint4 u[2]; } a;
    a.u[0] = *(const uint4*)p;
    a.u[1] = *(const uint4*)(p + 8);
    return a.v;
}

// B fragment from swizzled storage: per 32(K)x16(N) subtile, lane's 16
// elements are contiguous -> one 32B vector load.
__device__ __forceinline__ bf16x16 load_B_frag_sw(const bf16* smem) {
    int lane = threadIdx.x & 31;
    return *(const bf16x16*)(smem + lane * 16);
}

// B fragment where K is the contiguous axis: element(k,n)=smem[n*nstride+k],
// lane n=(l&15), K offset (l>>4)*16 -> contiguous 32B per lane.
__device__ __forceinline__ bf16x16 load_B_frag_kcontig(const bf16* smem, int n_stride) {
    int lane = threadIdx.x & 31;
    return *(const bf16x16*)(smem + (lane & 15) * n_stride + (lane >> 4) * 16);
}

// Swizzled B index: batch-local slot for (K=k, N=n) with 32-wide K blocks.
__device__ __forceinline__ int bswz_idx(int k, int n) {
    int kb = k >> 5, kin = k & 31;
    int khalf = kin >> 4, e = kin & 15;
    int nsub = n >> 4, nl = n & 15;
    return ((kb * 4 + nsub) * 32 + (khalf * 16 + nl)) * 16 + e;
}

#if HAVE_TDM
// ---------------------------------------------------------------------------
// Tensor Data Mover: 2D tile load (bf16) global -> LDS. D# per ISA 8.3/8.4.
// 6-arg builtin form: (g0, g1, g2, g3, g4, cpol).
// ---------------------------------------------------------------------------
__device__ __forceinline__ void tdm_load_2d_bf16(unsigned lds_addr, const void* gaddr,
                                                 unsigned tile_d0, unsigned tile_d1,
                                                 unsigned long stride_elems) {
    unsigned long ga = (unsigned long)gaddr;
    v4u g0;
    g0[0] = 1u;                                            // count=1 (valid, user)
    g0[1] = lds_addr;                                      // lds_addr [63:32]
    g0[2] = (unsigned)(ga & 0xFFFFFFFFu);                  // global_addr lo
    g0[3] = (unsigned)((ga >> 32) & 0x01FFFFFFu) | (2u << 30); // addr[56:32] | type=2
    const unsigned td0 = 0x00100000u, td1 = 0x00100000u;   // large tensor dims (in-bounds tiles)
    unsigned long s0 = stride_elems;
    v8i g1;
    g1[0] = (int)(1u << 16);                               // data_size=1 (2B), mask=0
    g1[1] = (int)((td0 & 0xFFFFu) << 16);                  // tensor_dim0 lo16
    g1[2] = (int)((td0 >> 16) | ((td1 & 0xFFFFu) << 16));  // dim0 hi | dim1 lo
    g1[3] = (int)((td1 >> 16) | (tile_d0 << 16));          // dim1 hi | tile_dim0
    g1[4] = (int)(tile_d1 & 0xFFFFu);                      // tile_dim1 | tile_dim2=0
    g1[5] = (int)(s0 & 0xFFFFFFFFu);                       // dim0_stride lo32
    g1[6] = (int)((s0 >> 32) & 0xFFFFu);                   // stride hi16 | dim1_stride lo=0
    g1[7] = 0;
    v4i z4 = {0, 0, 0, 0};
    v8i z8 = {0, 0, 0, 0, 0, 0, 0, 0};
    __builtin_amdgcn_tensor_load_to_lds(g0, g1, z4, z4, z8, 0);
}
#endif

// ---------------------------------------------------------------------------
// Kernel 1: LayerNorm over DIM=512; head-split to bf16 xh in B-swizzled order
// grid (Nn, Bx), block 256
// ---------------------------------------------------------------------------
__global__ void k_ln_split(const float* __restrict__ x, const float* __restrict__ gam,
                           const float* __restrict__ bet, bf16* __restrict__ xh) {
    int n = blockIdx.x, b = blockIdx.y, tid = threadIdx.x;
    const float* row = x + ((long)b * Nn + n) * Dm;
    __shared__ float red[256];
    float v0 = row[tid], v1 = row[tid + 256];
    red[tid] = v0 + v1;
    __syncthreads();
    for (int st = 128; st > 0; st >>= 1) { if (tid < st) red[tid] += red[tid + st]; __syncthreads(); }
    float mean = red[0] * (1.f / Dm);
    __syncthreads();
    float d0 = v0 - mean, d1 = v1 - mean;
    red[tid] = d0 * d0 + d1 * d1;
    __syncthreads();
    for (int st = 128; st > 0; st >>= 1) { if (tid < st) red[tid] += red[tid + st]; __syncthreads(); }
    float inv = rsqrtf(red[0] * (1.f / Dm) + 1e-5f);
    int c0 = tid, c1 = tid + 256;
    float t0f = d0 * inv * gam[c0] + bet[c0];
    float t1f = d1 * inv * gam[c1] + bet[c1];
    // token n is the K index of the B operand; dim-in-head is N
    {
        int h = c0 >> 6, d = c0 & 63;
        xh[((long)(b * Hh + h)) * (Nn * Dh) + bswz_idx(n, d)] = (bf16)t0f;
    }
    {
        int h = c1 >> 6, d = c1 & 63;
        xh[((long)(b * Hh + h)) * (Nn * Dh) + bswz_idx(n, d)] = (bf16)t1f;
    }
}

// ---------------------------------------------------------------------------
// Kernel: cast fp32 adjacency -> bf16 (layer 0 only), packed stores
// ---------------------------------------------------------------------------
__global__ void k_cast_adj(const float* __restrict__ src, bf16* __restrict__ dst, long n) {
    long i = (((long)blockIdx.x) * blockDim.x + threadIdx.x) * 4;
    if (i + 3 < n) {
        float4 v = *(const float4*)(src + i);
        union { bf16 h[4]; uint2 u; } p;
        p.h[0] = (bf16)v.x; p.h[1] = (bf16)v.y; p.h[2] = (bf16)v.z; p.h[3] = (bf16)v.w;
        *(uint2*)(dst + i) = p.u;
    } else {
        for (long j = i; j < n; ++j) dst[j] = (bf16)src[j];
    }
}

// ---------------------------------------------------------------------------
// Kernel 2/6: panel GEMM  C[N,64] = A[N,N] @ Bsw[N,64], batched over B*H.
// A streamed via TDM (double-buffered); B tiles are contiguous 4KB (swizzled).
// grid (Nn/128, Bx*Hh), block 256 (8 waves), wave -> 16x64 output.
// ---------------------------------------------------------------------------
__global__ void k_gemm_nx64(const bf16* __restrict__ A, const bf16* __restrict__ Bsw,
                            bf16* __restrict__ C) {
    int batch = blockIdx.y;
    int m0 = blockIdx.x * 128;
    A   += (long)batch * Nn * Nn;
    Bsw += (long)batch * Nn * Dh;
    C   += (long)batch * Nn * Dh;
    __shared__ __align__(32) bf16 sA[2][128 * 32];
    __shared__ __align__(32) bf16 sB[2][32 * 64];
    int tid = threadIdx.x, wave = tid >> 5, lane = tid & 31;

    auto loadB = [&](int buf, int kb) {
        *(uint4*)&sB[buf][tid * 8] = *(const uint4*)(Bsw + (long)kb * 2048 + tid * 8);
    };
#if HAVE_TDM
    auto loadA = [&](int buf, int kb) {
        if (wave == 0)
            tdm_load_2d_bf16((unsigned)(unsigned long)(void*)&sA[buf][0],
                             A + (long)m0 * Nn + kb * 32, 32u, 128u,
                             (unsigned long)Nn);
    };
#else
    auto loadA = [&](int buf, int kb) {
        int r = tid >> 1, c = (tid & 1) * 16;
        const uint4* s4 = (const uint4*)(A + (long)(m0 + r) * Nn + kb * 32 + c);
        uint4* d4 = (uint4*)&sA[buf][r * 32 + c];
        d4[0] = s4[0]; d4[1] = s4[1];
    };
#endif

    f32x8 acc[4];
#pragma unroll
    for (int t = 0; t < 4; ++t) acc[t] = zero8();

    loadA(0, 0);
    loadB(0, 0);
#if HAVE_TDM
    if (wave == 0) __builtin_amdgcn_s_wait_tensorcnt((short)0);
#endif
    __syncthreads();

    const int KB = Nn / 32;
    for (int kb = 0; kb < KB; ++kb) {
        int cur = kb & 1;
        if (kb + 1 < KB) {
            loadA(cur ^ 1, kb + 1);
            loadB(cur ^ 1, kb + 1);
        }
        bf16x16 af = load_A_frag(&sA[cur][(wave * 16) * 32], 32);
#pragma unroll
        for (int t = 0; t < 4; ++t)
            acc[t] = wmma_bf16(af, load_B_frag_sw(&sB[cur][t * 512]), acc[t]);
#if HAVE_TDM
        if (wave == 0 && kb + 1 < KB) __builtin_amdgcn_s_wait_tensorcnt((short)0);
#endif
        __syncthreads();
    }
    int nl = lane & 15, mb = (lane >> 4) * 8;
#pragma unroll
    for (int t = 0; t < 4; ++t)
#pragma unroll
        for (int r = 0; r < 8; ++r)
            C[(long)(m0 + wave * 16 + mb + r) * Dh + t * 16 + nl] = (bf16)acc[t][r];
}

// ---------------------------------------------------------------------------
// Kernel 3: q = ax @ Wq, k = ax @ Wk (K=64). Weights fp32 -> swizzled bf16 LDS.
// grid (Nn/128, Bx*Hh)
// ---------------------------------------------------------------------------
__global__ void k_qk(const bf16* __restrict__ ax, const float* __restrict__ Wq,
                     const float* __restrict__ Wk, bf16* __restrict__ q,
                     bf16* __restrict__ kk) {
    int batch = blockIdx.y;
    int m0 = blockIdx.x * 128;
    long off = (long)batch * Nn * Dh;
    ax += off; q += off; kk += off;
    __shared__ __align__(32) bf16 sW[2][Dh * Dh];
    __shared__ __align__(32) bf16 sA[128 * Dh];
    int tid = threadIdx.x, wave = tid >> 5, lane = tid & 31;
    for (int idx = tid; idx < Dh * Dh; idx += 256) {
        int e = idx & 15, L = (idx >> 4) & 31, nsub = (idx >> 9) & 3, kb = idx >> 11;
        int k = kb * 32 + (L >> 4) * 16 + e;
        int n = nsub * 16 + (L & 15);
        sW[0][idx] = (bf16)Wq[k * Dh + n];
        sW[1][idx] = (bf16)Wk[k * Dh + n];
    }
    {   // A tile 128x64 row-major
        int r = tid >> 1, c = (tid & 1) * 32;
        const uint4* s4 = (const uint4*)(ax + (long)(m0 + r) * Dh + c);
        uint4* d4 = (uint4*)&sA[r * Dh + c];
        d4[0] = s4[0]; d4[1] = s4[1]; d4[2] = s4[2]; d4[3] = s4[3];
    }
    __syncthreads();
    f32x8 acc[2][4];
#pragma unroll
    for (int w = 0; w < 2; ++w)
#pragma unroll
        for (int t = 0; t < 4; ++t) acc[w][t] = zero8();
#pragma unroll
    for (int kb = 0; kb < 2; ++kb) {
        bf16x16 af = load_A_frag(&sA[(wave * 16) * Dh + kb * 32], Dh);
#pragma unroll
        for (int w = 0; w < 2; ++w)
#pragma unroll
            for (int t = 0; t < 4; ++t)
                acc[w][t] = wmma_bf16(af, load_B_frag_sw(&sW[w][(kb * 4 + t) * 512]), acc[w][t]);
    }
    int nl = lane & 15, mb = (lane >> 4) * 8;
#pragma unroll
    for (int t = 0; t < 4; ++t)
#pragma unroll
        for (int r = 0; r < 8; ++r) {
            long idx = (long)(m0 + wave * 16 + mb + r) * Dh + t * 16 + nl;
            q[idx]  = (bf16)acc[0][t][r];
            kk[idx] = (bf16)acc[1][t][r];
        }
}

// ---------------------------------------------------------------------------
// Kernel 4: s = leakyrelu(scale * q @ k^T), bf16 out. 128(l) x 64(t) per block.
// grid (Nn/128, Nn/64, Bx*Hh)
// ---------------------------------------------------------------------------
__global__ void k_attn(const bf16* __restrict__ q, const bf16* __restrict__ k,
                       bf16* __restrict__ s) {
    int batch = blockIdx.z;
    int m0 = blockIdx.x * 128, t0 = blockIdx.y * 64;
    q += (long)batch * Nn * Dh;
    k += (long)batch * Nn * Dh;
    s += (long)batch * Nn * Nn;
    __shared__ __align__(32) bf16 sQ[128 * Dh];
    __shared__ __align__(32) bf16 sK[64 * Dh];
    int tid = threadIdx.x, wave = tid >> 5, lane = tid & 31;
    {
        int r = tid >> 1, c = (tid & 1) * 32;
        const uint4* s4 = (const uint4*)(q + (long)(m0 + r) * Dh + c);
        uint4* d4 = (uint4*)&sQ[r * Dh + c];
        d4[0] = s4[0]; d4[1] = s4[1]; d4[2] = s4[2]; d4[3] = s4[3];
    }
    {
        int r = tid >> 2, c = (tid & 3) * 16;
        const uint4* s4 = (const uint4*)(k + (long)(t0 + r) * Dh + c);
        uint4* d4 = (uint4*)&sK[r * Dh + c];
        d4[0] = s4[0]; d4[1] = s4[1];
    }
    __syncthreads();
    f32x8 acc[4];
#pragma unroll
    for (int t = 0; t < 4; ++t) acc[t] = zero8();
#pragma unroll
    for (int k0 = 0; k0 < Dh; k0 += 32) {
        bf16x16 af = load_A_frag(&sQ[(wave * 16) * Dh + k0], Dh);
#pragma unroll
        for (int t = 0; t < 4; ++t)   // B(k=d, n=tcol) = sK[tcol*Dh + d]: K contiguous
            acc[t] = wmma_bf16(af, load_B_frag_kcontig(&sK[(t * 16) * Dh + k0], Dh), acc[t]);
    }
    int nl = lane & 15, mb = (lane >> 4) * 8;
#pragma unroll
    for (int t = 0; t < 4; ++t)
#pragma unroll
        for (int r = 0; r < 8; ++r) {
            float v = acc[t][r] * ATTN_SCALE;
            v = (v >= 0.f) ? v : SLOPE * v;
            int row = m0 + wave * 16 + mb + r, col = t0 + t * 16 + nl;
            s[(long)row * Nn + col] = (bf16)v;
        }
}

// ---------------------------------------------------------------------------
// Kernel 5: head mixing (Lam), optional layer-0 mask, row softmax -> rep bf16
// grid (Nn, Bx), block 256. Thread owns 4 contiguous columns in registers.
// ---------------------------------------------------------------------------
__global__ void k_mix_softmax(const bf16* __restrict__ s, const float* __restrict__ Lam,
                              const float* __restrict__ maskAdj, bf16* __restrict__ rep) {
    int l = blockIdx.x, b = blockIdx.y, tid = threadIdx.x;
    __shared__ __align__(16) bf16 sRow[Hh][Nn];
    __shared__ float red[256];
    for (int idx = tid; idx < (Hh * Nn) / 8; idx += 256) {
        int h = idx >> 7;
        int t8 = (idx & 127) * 8;
        *(uint4*)&sRow[h][t8] = *(const uint4*)&s[(((long)(b * Hh + h)) * Nn + l) * Nn + t8];
    }
    __syncthreads();
    int t0 = tid * 4;
    for (int g = 0; g < Hh; ++g) {
        float lam[Hh];
#pragma unroll
        for (int h = 0; h < Hh; ++h) lam[h] = Lam[g * Hh + h];
        long rbase = (((long)(b * Hh + g)) * Nn + l) * Nn;
        float m[4];
#pragma unroll
        for (int j = 0; j < 4; ++j) {
            float a = 0.f;
#pragma unroll
            for (int h = 0; h < Hh; ++h) a += lam[h] * (float)sRow[h][t0 + j];
            m[j] = a;
        }
        if (maskAdj) {
            float4 a4 = *(const float4*)&maskAdj[rbase + t0];
            if (!(a4.x > 0.f)) m[0] = -1.0e30f;
            if (!(a4.y > 0.f)) m[1] = -1.0e30f;
            if (!(a4.z > 0.f)) m[2] = -1.0e30f;
            if (!(a4.w > 0.f)) m[3] = -1.0e30f;
        }
        float lmax = fmaxf(fmaxf(m[0], m[1]), fmaxf(m[2], m[3]));
        red[tid] = lmax; __syncthreads();
        for (int st = 128; st > 0; st >>= 1) { if (tid < st) red[tid] = fmaxf(red[tid], red[tid + st]); __syncthreads(); }
        float rmax = red[0]; __syncthreads();
        float e[4], lsum = 0.f;
#pragma unroll
        for (int j = 0; j < 4; ++j) { e[j] = __expf(m[j] - rmax); lsum += e[j]; }
        red[tid] = lsum; __syncthreads();
        for (int st = 128; st > 0; st >>= 1) { if (tid < st) red[tid] += red[tid + st]; __syncthreads(); }
        float inv = 1.f / red[0]; __syncthreads();
        union { bf16 h[4]; uint2 u; } o;
#pragma unroll
        for (int j = 0; j < 4; ++j) o.h[j] = (bf16)(e[j] * inv);
        *(uint2*)&rep[rbase + t0] = o.u;
    }
}

// ---------------------------------------------------------------------------
// Kernel 7: v = vtmp @ Wv, GELU(exact), head-merge to gact[B,N,512] bf16
// grid (Nn/128, Bx*Hh)
// ---------------------------------------------------------------------------
__global__ void k_v_gelu(const bf16* __restrict__ vt, const float* __restrict__ Wv,
                         bf16* __restrict__ gact) {
    int batch = blockIdx.y;
    int b = batch >> 3, h = batch & 7;
    int m0 = blockIdx.x * 128;
    vt += (long)batch * Nn * Dh;
    __shared__ __align__(32) bf16 sW[Dh * Dh];
    __shared__ __align__(32) bf16 sA[128 * Dh];
    int tid = threadIdx.x, wave = tid >> 5, lane = tid & 31;
    for (int idx = tid; idx < Dh * Dh; idx += 256) {
        int e = idx & 15, L = (idx >> 4) & 31, nsub = (idx >> 9) & 3, kb = idx >> 11;
        int k = kb * 32 + (L >> 4) * 16 + e;
        int n = nsub * 16 + (L & 15);
        sW[idx] = (bf16)Wv[k * Dh + n];
    }
    {
        int r = tid >> 1, c = (tid & 1) * 32;
        const uint4* s4 = (const uint4*)(vt + (long)(m0 + r) * Dh + c);
        uint4* d4 = (uint4*)&sA[r * Dh + c];
        d4[0] = s4[0]; d4[1] = s4[1]; d4[2] = s4[2]; d4[3] = s4[3];
    }
    __syncthreads();
    f32x8 acc[4];
#pragma unroll
    for (int t = 0; t < 4; ++t) acc[t] = zero8();
#pragma unroll
    for (int kb = 0; kb < 2; ++kb) {
        bf16x16 af = load_A_frag(&sA[(wave * 16) * Dh + kb * 32], Dh);
#pragma unroll
        for (int t = 0; t < 4; ++t)
            acc[t] = wmma_bf16(af, load_B_frag_sw(&sW[(kb * 4 + t) * 512]), acc[t]);
    }
    int nl = lane & 15, mb = (lane >> 4) * 8;
#pragma unroll
    for (int t = 0; t < 4; ++t)
#pragma unroll
        for (int r = 0; r < 8; ++r) {
            float v = acc[t][r];
            v = 0.5f * v * (1.f + erff(v * 0.70710678118654752f));
            int row = m0 + wave * 16 + mb + r;
            gact[((long)b * Nn + row) * Dm + h * Dh + t * 16 + nl] = (bf16)v;
        }
}

// ---------------------------------------------------------------------------
// Kernel 8: out = gact @ Wp^T + x  (M=B*N=2048, K=512, N=512), fp32 out
// grid ((Bx*Nn)/128, Dm/64)
// ---------------------------------------------------------------------------
__global__ void k_out_res(const bf16* __restrict__ gact, const float* __restrict__ Wp,
                          const float* __restrict__ xin, float* __restrict__ xout) {
    int m0 = blockIdx.x * 128, n0 = blockIdx.y * 64;
    int tid = threadIdx.x, wave = tid >> 5, lane = tid & 31;
    __shared__ __align__(32) bf16 sG[128 * 32];
    __shared__ __align__(32) bf16 sW[64 * 32];   // [n][k] -> K contiguous
    f32x8 acc[4];
#pragma unroll
    for (int t = 0; t < 4; ++t) acc[t] = zero8();

    for (int k0 = 0; k0 < Dm; k0 += 32) {
        __syncthreads();
        {
            int r = tid >> 1, c = (tid & 1) * 16;
            const uint4* s4 = (const uint4*)(gact + (long)(m0 + r) * Dm + k0 + c);
            uint4* d4 = (uint4*)&sG[r * 32 + c];
            d4[0] = s4[0]; d4[1] = s4[1];
        }
        for (int idx = tid; idx < 64 * 32; idx += 256) {
            int nn = idx >> 5, kk = idx & 31;
            sW[idx] = (bf16)Wp[(long)(n0 + nn) * Dm + k0 + kk];  // B(k,n) = Wp[n,k]
        }
        __syncthreads();
        bf16x16 af = load_A_frag(&sG[(wave * 16) * 32], 32);
#pragma unroll
        for (int t = 0; t < 4; ++t)
            acc[t] = wmma_bf16(af, load_B_frag_kcontig(&sW[(t * 16) * 32], 32), acc[t]);
    }
    int nl = lane & 15, mb = (lane >> 4) * 8;
#pragma unroll
    for (int t = 0; t < 4; ++t)
#pragma unroll
        for (int r = 0; r < 8; ++r) {
            int row = m0 + wave * 16 + mb + r, col = n0 + t * 16 + nl;
            xout[(long)row * Dm + col] = acc[t][r] + xin[(long)row * Dm + col];
        }
}

// ---------------------------------------------------------------------------
// Host orchestration
// ---------------------------------------------------------------------------
extern "C" void kernel_launch(void* const* d_in, const int* in_sizes, int n_in,
                              void* d_out, int out_size, void* d_ws, size_t ws_size,
                              hipStream_t stream) {
    const float* x_in   = (const float*)d_in[0];
    const float* adj_in = (const float*)d_in[1];
    const float* ln_g   = (const float*)d_in[2];
    const float* ln_b   = (const float*)d_in[3];
    const float* Wq     = (const float*)d_in[4];
    const float* Wk     = (const float*)d_in[5];
    const float* Wv     = (const float*)d_in[6];
    const float* Lam    = (const float*)d_in[7];
    const float* Wp     = (const float*)d_in[8];

    char* w = (char*)d_ws;
    auto carve = [&](size_t bytes) -> void* {
        void* p = (void*)w;
        w += (bytes + 255) & ~(size_t)255;
        return p;
    };
    const size_t szX   = (size_t)Bx * Nn * Dm * sizeof(float);        // 4 MB
    const size_t szXH  = (size_t)Bx * Hh * Nn * Dh * sizeof(bf16);    // 2 MB
    const size_t szAdj = (size_t)Bx * Hh * Nn * Nn * sizeof(bf16);    // 32 MB
    float* x0  = (float*)carve(szX);
    float* x1  = (float*)carve(szX);
    bf16* xh   = (bf16*)carve(szXH);
    bf16* adjA = (bf16*)carve(szAdj);
    bf16* adjB = (bf16*)carve(szAdj);
    bf16* sbuf = (bf16*)carve(szAdj);
    bf16* ax   = (bf16*)carve(szXH);
    bf16* qb   = (bf16*)carve(szXH);
    bf16* kb   = (bf16*)carve(szXH);
    bf16* vtmp = (bf16*)carve(szXH);
    bf16* gact = (bf16*)carve(szX / 2);

    (void)hipMemcpyAsync(x0, x_in, szX, hipMemcpyDeviceToDevice, stream);
    {
        long n = (long)Bx * Hh * Nn * Nn;
        long blocks = (n / 4 + 255) / 256;
        k_cast_adj<<<dim3((unsigned)blocks), 256, 0, stream>>>(adj_in, adjA, n);
    }

    float* xcur = x0;
    float* xnext = x1;
    bf16* adjCur = adjA;
    bf16* adjNext = adjB;

    for (int i = 0; i < Ll; ++i) {
        k_ln_split<<<dim3(Nn, Bx), 256, 0, stream>>>(
            xcur, ln_g + (long)i * Dm, ln_b + (long)i * Dm, xh);
        k_gemm_nx64<<<dim3(Nn / 128, Bx * Hh), 256, 0, stream>>>(adjCur, xh, ax);
        k_qk<<<dim3(Nn / 128, Bx * Hh), 256, 0, stream>>>(
            ax, Wq + (long)i * Dh * Dh, Wk + (long)i * Dh * Dh, qb, kb);
        k_attn<<<dim3(Nn / 128, Nn / 64, Bx * Hh), 256, 0, stream>>>(qb, kb, sbuf);
        k_mix_softmax<<<dim3(Nn, Bx), 256, 0, stream>>>(
            sbuf, Lam + (long)i * Hh * Hh, (i == 0) ? adj_in : nullptr, adjNext);
        k_gemm_nx64<<<dim3(Nn / 128, Bx * Hh), 256, 0, stream>>>(adjNext, xh, vtmp);
        k_v_gelu<<<dim3(Nn / 128, Bx * Hh), 256, 0, stream>>>(
            vtmp, Wv + (long)i * Dh * Dh, gact);
        k_out_res<<<dim3((Bx * Nn) / 128, Dm / 64), 256, 0, stream>>>(
            gact, Wp + (long)i * Dm * Dm, xcur, xnext);

        float* tx = xcur; xcur = xnext; xnext = tx;
        bf16* ta = adjCur; adjCur = adjNext; adjNext = ta;
    }

    (void)hipMemcpyAsync(d_out, xcur, szX, hipMemcpyDeviceToDevice, stream);
}
